// AutoEncoder_20822001451042
// MI455X (gfx1250) — matrix operand order repaired and checked by reading, not compiled, not run on hardware
//
#include <hip/hip_runtime.h>
#include <stdint.h>

// ---------------------------------------------------------------------------
// GCN autoencoder: x@W -> norm scatter-add -> BN -> ReLU -> z@z.T
// N=12000, E=384000, F_IN=256, F_HID=128. Output: [N,N] f32 (576 MB).
// Dominant cost is the Gram matrix -> bf16 WMMA (hi/lo split for accuracy).
// ---------------------------------------------------------------------------

typedef __attribute__((ext_vector_type(16))) __bf16          v16bf;
typedef __attribute__((ext_vector_type(8)))  float           v8f;
typedef __attribute__((ext_vector_type(8)))  unsigned short  us8;
typedef __attribute__((ext_vector_type(16))) unsigned short  us16;

#define DEVFN __device__ __forceinline__

constexpr int FIN = 256;
constexpr int FH  = 128;
constexpr float BN_EPS = 1e-5f;

// ---------------- small helpers ----------------

DEVFN unsigned short f32_to_bf16_rne(float f) {
  unsigned int u = __float_as_uint(f);
  unsigned int r = u + 0x7FFFu + ((u >> 16) & 1u);
  return (unsigned short)(r >> 16);
}
DEVFN float bf16_to_f32(unsigned short h) {
  return __uint_as_float(((unsigned int)h) << 16);
}

// ---------------- kernel 1: degree (self-loop => init 1) ----------------

__global__ void k_deg_init(float* deg, int n) {
  int i = blockIdx.x * blockDim.x + threadIdx.x;
  if (i < n) deg[i] = 1.0f;
}

__global__ void k_deg_count(const int* __restrict__ ei, float* deg, int E) {
  int e = blockIdx.x * blockDim.x + threadIdx.x;
  if (e < E) atomicAdd(&deg[ei[E + e]], 1.0f);   // dst row of edge_index
}

__global__ void k_dinv(float* deg, int n) {
  int i = blockIdx.x * blockDim.x + threadIdx.x;
  if (i < n) deg[i] = rsqrtf(deg[i]);
}

// ---------------- kernel 2: xw = x @ W  (f32, tiny GEMM) ----------------
// block = 128 threads, one block per row; thread c does the dot for column c.

__global__ void __launch_bounds__(FH) k_xw(const float* __restrict__ x,
                                           const float* __restrict__ W,
                                           float* __restrict__ xw) {
  const int row = blockIdx.x;
  const int c   = threadIdx.x;
  const float* xr = x + (size_t)row * FIN;
  float acc = 0.0f;
#pragma unroll 4
  for (int k = 0; k < FIN; ++k)
    acc = fmaf(xr[k], W[(size_t)k * FH + c], acc);
  xw[(size_t)row * FH + c] = acc;
}

// ---------------- kernel 3: h = b (bias pre-load) ----------------

__global__ void k_hinit(float* __restrict__ h, const float* __restrict__ b, int total) {
  int idx = blockIdx.x * blockDim.x + threadIdx.x;
  if (idx < total) h[idx] = b[idx & (FH - 1)];
}

// ---------------- kernel 4: normalized scatter-add ----------------
// one block (128 threads) per message; messages [0,E) are edges, [E,E+N) self-loops.

__global__ void __launch_bounds__(FH) k_scatter(const int* __restrict__ ei,
                                                const float* __restrict__ dinv,
                                                const float* __restrict__ xw,
                                                float* __restrict__ h, int E) {
  const int m = blockIdx.x;
  const int c = threadIdx.x;
  int s, d;
  if (m < E) { s = ei[m]; d = ei[E + m]; }
  else       { s = d = m - E; }
  const float nrm = dinv[s] * dinv[d];
  atomicAdd(&h[(size_t)d * FH + c], xw[(size_t)s * FH + c] * nrm);
}

// ---------------- kernel 5: BN column stats -> scale/shift ----------------
// one block per feature column (128 blocks x 256 threads).

__global__ void __launch_bounds__(256) k_bnstats(const float* __restrict__ h,
                                                 const float* __restrict__ gamma,
                                                 const float* __restrict__ beta,
                                                 float* __restrict__ ss, int n) {
  const int c = blockIdx.x;
  const int t = threadIdx.x;
  float s1 = 0.0f, s2 = 0.0f;
  for (int r = t; r < n; r += 256) {
    float v = h[(size_t)r * FH + c];
    s1 += v;
    s2 += v * v;
  }
  __shared__ float sh1[256], sh2[256];
  sh1[t] = s1; sh2[t] = s2;
  __syncthreads();
  for (int ofs = 128; ofs > 0; ofs >>= 1) {
    if (t < ofs) { sh1[t] += sh1[t + ofs]; sh2[t] += sh2[t + ofs]; }
    __syncthreads();
  }
  if (t == 0) {
    float inv_n = 1.0f / (float)n;
    float mean = sh1[0] * inv_n;
    float var  = sh2[0] * inv_n - mean * mean;           // biased variance
    float sc   = gamma[c] * rsqrtf(var + BN_EPS);
    ss[c]      = sc;
    ss[FH + c] = beta[c] - mean * sc;
  }
}

// ---------------- kernel 6: normalize + ReLU + bf16 hi/lo split ----------------

__global__ void k_norm(const float* __restrict__ h, const float* __restrict__ ss,
                       unsigned short* __restrict__ zhi, unsigned short* __restrict__ zlo,
                       int total) {
  int idx = blockIdx.x * blockDim.x + threadIdx.x;
  if (idx >= total) return;
  int c = idx & (FH - 1);
  float v = fmaf(h[idx], ss[c], ss[FH + c]);
  v = fmaxf(v, 0.0f);
  unsigned short hi = f32_to_bf16_rne(v);
  float residual = v - bf16_to_f32(hi);
  zhi[idx] = hi;
  zlo[idx] = f32_to_bf16_rne(residual);
}

// ---------------- kernel 7: out = z @ z.T via bf16 WMMA ----------------
// A frag (16x32 bf16, M-major): lane l (row M=l&15) holds two contiguous 8-elem
//   K-chunks at k0 and k0+16 with k0 = kstep*32 + (l<16 ? 0 : 8).
// B frag (32x16 bf16, z.T column = z row): lane l (col N=l&15) holds two
//   contiguous 8-elem K-chunks at k0 and k0+16, k0 = kstep*32 + (l<16 ? 0 : 8).
// Both are straight 16B loads from rows of z -> no transpose needed.

DEVFN v16bf load_a_frag(const unsigned short* __restrict__ z, int row, int kk) {
  const unsigned short* p = z + (size_t)row * FH + kk;
  us8 c0 = *(const us8*)(p);
  us8 c1 = *(const us8*)(p + 16);
  us16 f = __builtin_shufflevector(c0, c1, 0,1,2,3,4,5,6,7,8,9,10,11,12,13,14,15);
  return __builtin_bit_cast(v16bf, f);
}

DEVFN v16bf load_b_frag(const unsigned short* __restrict__ z, int col, int kb) {
  const unsigned short* p = z + (size_t)col * FH + kb;
  us8 c0 = *(const us8*)(p);
  us8 c1 = *(const us8*)(p + 16);
  us16 f = __builtin_shufflevector(c0, c1, 0,1,2,3,4,5,6,7,8,9,10,11,12,13,14,15);
  return __builtin_bit_cast(v16bf, f);
}

__global__ void __launch_bounds__(128) k_gram(const unsigned short* __restrict__ zhi,
                                              const unsigned short* __restrict__ zlo,
                                              float* __restrict__ out, int n) {
  const int lane   = threadIdx.x & 31;
  const int wave   = threadIdx.x >> 5;      // 4 waves / block
  const int wm     = wave >> 1;             // 0..1
  const int wn     = wave & 1;              // 0..1
  const int row0   = blockIdx.y * 64 + wm * 32;   // wave tile: 32 rows
  const int col0   = blockIdx.x * 128 + wn * 64;  //            64 cols
  const int lm     = lane & 15;
  const bool hiH   = lane >= 16;

  v8f acc[2][4];
#pragma unroll
  for (int mt = 0; mt < 2; ++mt)
#pragma unroll
    for (int nt = 0; nt < 4; ++nt)
      acc[mt][nt] = (v8f){0.f, 0.f, 0.f, 0.f, 0.f, 0.f, 0.f, 0.f};

#pragma unroll
  for (int ks = 0; ks < 4; ++ks) {          // K = 128 in steps of 32
    const int kkA = ks * 32 + (hiH ? 8 : 0);
    const int kkB = ks * 32 + (hiH ? 8 : 0);

    v16bf ah[2], al[2], bh[4], bl[4];
#pragma unroll
    for (int mt = 0; mt < 2; ++mt) {
      int r = row0 + mt * 16 + lm;
      if (r >= n) r = n - 1;                // clamp loads; stores guarded below
      ah[mt] = load_a_frag(zhi, r, kkA);
      al[mt] = load_a_frag(zlo, r, kkA);
    }
#pragma unroll
    for (int nt = 0; nt < 4; ++nt) {
      int c = col0 + nt * 16 + lm;
      if (c >= n) c = n - 1;
      bh[nt] = load_b_frag(zhi, c, kkB);
      bl[nt] = load_b_frag(zlo, c, kkB);
    }

#pragma unroll
    for (int mt = 0; mt < 2; ++mt)
#pragma unroll
      for (int nt = 0; nt < 4; ++nt) {
        acc[mt][nt] = __builtin_amdgcn_wmma_f32_16x16x32_bf16(
            false, ah[mt], false, bh[nt], (short)0, acc[mt][nt], false, false);
        acc[mt][nt] = __builtin_amdgcn_wmma_f32_16x16x32_bf16(
            false, ah[mt], false, bl[nt], (short)0, acc[mt][nt], false, false);
        acc[mt][nt] = __builtin_amdgcn_wmma_f32_16x16x32_bf16(
            false, al[mt], false, bh[nt], (short)0, acc[mt][nt], false, false);
      }
  }

  // C/D layout: element r -> row = r + (lane>=16 ? 8:0), col = lane&15
#pragma unroll
  for (int mt = 0; mt < 2; ++mt) {
#pragma unroll
    for (int nt = 0; nt < 4; ++nt) {
      const int c = col0 + nt * 16 + lm;
      if (c >= n) continue;
#pragma unroll
      for (int r = 0; r < 8; ++r) {
        const int rr = row0 + mt * 16 + r + (hiH ? 8 : 0);
        if (rr < n) out[(size_t)rr * (size_t)n + c] = acc[mt][nt][r];
      }
    }
  }
}

// ---------------------------------------------------------------------------
// host launcher
// ---------------------------------------------------------------------------

static inline size_t align256(size_t v) { return (v + 255) & ~(size_t)255; }

extern "C" void kernel_launch(void* const* d_in, const int* in_sizes, int n_in,
                              void* d_out, int out_size, void* d_ws, size_t ws_size,
                              hipStream_t stream) {
  const float* x     = (const float*)d_in[0];
  const int*   ei    = (const int*)  d_in[1];   // [2,E] flat: src then dst
  const float* W     = (const float*)d_in[2];
  const float* b     = (const float*)d_in[3];
  const float* gamma = (const float*)d_in[4];
  const float* beta  = (const float*)d_in[5];

  const int n = in_sizes[0] / FIN;   // 12000
  const int E = in_sizes[1] / 2;     // 384000

  char* ws = (char*)d_ws;
  size_t off = 0;
  float* deg = (float*)(ws + off);              off += align256((size_t)n * 4);
  float* xw  = (float*)(ws + off);              off += align256((size_t)n * FH * 4);
  float* h   = (float*)(ws + off);              off += align256((size_t)n * FH * 4);
  float* ss  = (float*)(ws + off);              off += align256(2 * FH * 4);
  unsigned short* zhi = (unsigned short*)(ws + off); off += align256((size_t)n * FH * 2);
  unsigned short* zlo = (unsigned short*)(ws + off); off += align256((size_t)n * FH * 2);
  (void)ws_size; (void)n_in;

  float* out = (float*)d_out;
  const int total = n * FH;

  k_deg_init <<<(n + 255) / 256, 256, 0, stream>>>(deg, n);
  k_deg_count<<<(E + 255) / 256, 256, 0, stream>>>(ei, deg, E);
  k_dinv     <<<(n + 255) / 256, 256, 0, stream>>>(deg, n);

  k_xw       <<<n, FH, 0, stream>>>(x, W, xw);
  k_hinit    <<<(total + 255) / 256, 256, 0, stream>>>(h, b, total);
  k_scatter  <<<E + n, FH, 0, stream>>>(ei, deg, xw, h, E);

  k_bnstats  <<<FH, 256, 0, stream>>>(h, gamma, beta, ss, n);
  k_norm     <<<(total + 255) / 256, 256, 0, stream>>>(h, ss, zhi, zlo, total);

  dim3 grid((n + 127) / 128, (n + 63) / 64);
  k_gram     <<<grid, 128, 0, stream>>>(zhi, zlo, out, n);
}
